// GetRoisFeature_56942676410702
// MI455X (gfx1250) — matrix operand order repaired and checked by reading, not compiled, not run on hardware
//
#include <hip/hip_runtime.h>
#include <hip/hip_bf16.h>

typedef __attribute__((ext_vector_type(16))) _Float16 v16h;
typedef __attribute__((ext_vector_type(8)))  float    v8f;

#define R_TOT 4096
#define NB    8
#define CCH   64
#define HH    64
#define WW    64
#define MAXT  264   // sum_b ceil(cnt_b/16) <= 4096/16 + 8

#if defined(__has_builtin) && __has_builtin(__builtin_amdgcn_sched_barrier)
#define SCHED_BARRIER() __builtin_amdgcn_sched_barrier(0)
#else
#define SCHED_BARRIER() do {} while (0)
#endif

union AU { v16h v; uint u[8]; _Float16 h[16]; };

__device__ __forceinline__ float tent_int(float t) {
    // antiderivative of max(0, 1-|t|), clipped to [-1,1]
    t = fminf(fmaxf(t, -1.0f), 1.0f);
    float lo = 0.5f * (t + 1.0f) * (t + 1.0f);
    float hi = 1.0f - 0.5f * (1.0f - t) * (1.0f - t);
    return (t < 0.0f) ? lo : hi;
}

__global__ void k_zero(int* __restrict__ counts) {
    if (threadIdx.x < NB) counts[threadIdx.x] = 0;
}

// fp32 feature -> fp16 copy (L2-resident WMMA operand source)
__global__ void k_cvt(const float* __restrict__ f, _Float16* __restrict__ h, int n4) {
    int i = blockIdx.x * blockDim.x + threadIdx.x;
    if (i >= n4) return;
    float4 v = ((const float4*)f)[i];
    union { _Float16 x[4]; uint2 u; } p;
    p.x[0] = (_Float16)v.x; p.x[1] = (_Float16)v.y;
    p.x[2] = (_Float16)v.z; p.x[3] = (_Float16)v.w;
    ((uint2*)h)[i] = p.u;
}

// Per-ROI telescoped axis weights Wx/Wy, 1/(49*area), and batch binning.
__global__ void k_weights(const float* __restrict__ rois,
                          float* __restrict__ Wx, float* __restrict__ Wy,
                          float* __restrict__ inv,
                          int* __restrict__ counts, int* __restrict__ lists) {
    int r = blockIdx.x * blockDim.x + threadIdx.x;
    if (r >= R_TOT) return;
    // reference row r = transpose(rois,(1,0,..)): r = b_outer*512 + n
    int bo = r >> 9, n = r & 511;
    const float* p = rois + (size_t)(n * NB + bo) * 5;
    int   b  = (int)p[0];
    float x1 = p[1], y1 = p[2], x2 = p[3], y2 = p[4];
    #pragma unroll 4
    for (int j = 0; j < WW; ++j) {
        float fj = (float)j;
        Wx[r * WW + j] = tent_int(x2 - fj) - tent_int(x1 - fj);
        Wy[r * HH + j] = tent_int(y2 - fj) - tent_int(y1 - fj);
    }
    float bw = (x2 - x1) * (1.0f / 7.0f);
    float bh = (y2 - y1) * (1.0f / 7.0f);
    float area = fmaxf(bw * bh, 0.0f);
    inv[r] = (area > 0.0f) ? 1.0f / (49.0f * fmaxf(area, 1e-12f)) : 0.0f;
    int pos = atomicAdd(&counts[b], 1);
    lists[b * R_TOT + pos] = r;
}

// Build tile table: groups of 16 ROIs sharing a feature batch (pad w/ last).
__global__ void k_tiles(const int* __restrict__ counts, const int* __restrict__ lists,
                        int* __restrict__ tileRoi, int* __restrict__ tileBatch) {
    if (threadIdx.x != 0 || blockIdx.x != 0) return;
    int t = 0;
    for (int b = 0; b < NB; ++b) {
        int cnt = counts[b];
        int nt = (cnt + 15) >> 4;
        for (int ti = 0; ti < nt; ++ti) {
            for (int e = 0; e < 16; ++e) {
                int idx = ti * 16 + e;
                if (idx >= cnt) idx = cnt - 1;     // pad: duplicate writes of identical values
                tileRoi[t * 16 + e] = lists[b * R_TOT + idx];
            }
            tileBatch[t++] = b;
        }
    }
    for (; t < MAXT; ++t) tileBatch[t] = -1;
}

// Main kernel: one workgroup (4 waves) per 16-ROI tile; wave owns 16 channels.
// Stage1 (WMMA f16->f32): S[i,r] = sum_j F[b,c,i,j] * Wx[r,j]
// Stage2 (VALU on D regs): out[r,c] = inv_r * sum_i Wy[r,i] * S[i,r]
__global__ void __launch_bounds__(128)
k_pool(const _Float16* __restrict__ feat16,
       const float* __restrict__ Wx, const float* __restrict__ Wy,
       const float* __restrict__ inv,
       const int* __restrict__ tileRoi, const int* __restrict__ tileBatch,
       float* __restrict__ out) {
    int blk = blockIdx.x;
    int tb = tileBatch[blk];
    if (tb < 0) return;                 // uniform per block: EXEC all-1s at WMMA

    int tid  = threadIdx.x;
    int lane = tid & 31;
    int wave = tid >> 5;
    int nl   = lane & 15;               // B/D column index -> ROI slot
    int half = lane >> 4;               // lane half

    int   roi  = tileRoi[blk * 16 + nl];
    float invr = inv[roi];

    // B operand (Wxᵀ, 32x16 f16): lane half h holds K = j0 + h*8 + {0..7, 16..23}
    AU bm[2];
    #pragma unroll
    for (int ks = 0; ks < 2; ++ks) {
        const float* wp = Wx + roi * WW + ks * 32 + half * 8;
        #pragma unroll
        for (int e = 0; e < 16; ++e) bm[ks].h[e] = (_Float16)wp[(e < 8) ? e : e + 8];
    }

    // Wy values matching D layout: D vgpr g holds row i = it*16 + half*8 + g
    float wyr[32];
    #pragma unroll
    for (int it = 0; it < 4; ++it)
        #pragma unroll
        for (int g = 0; g < 8; ++g)
            wyr[it * 8 + g] = Wy[roi * HH + it * 16 + half * 8 + g];

    const _Float16* fb = feat16 + (size_t)tb * CCH * HH * WW;

    for (int cl = 0; cl < 16; ++cl) {
        int c = wave * 16 + cl;
        const _Float16* fc = fb + (size_t)c * HH * WW;

        // Stage all 8 A-tiles for this channel up front (16 global_load_b128).
        AU a[8];
        #pragma unroll
        for (int it = 0; it < 4; ++it) {
            #pragma unroll
            for (int ks = 0; ks < 2; ++ks) {
                // A operand (F, 16x32 f16): per half, K groups {0..7} and {16..23}
                const _Float16* ap = fc + (it * 16 + nl) * WW + ks * 32 + half * 8;
                uint4 g0 = *(const uint4*)(ap);        // K = base+0..7   (16B aligned)
                uint4 g1 = *(const uint4*)(ap + 16);   // K = base+16..23 (16B aligned)
                int s = it * 2 + ks;
                a[s].u[0] = g0.x; a[s].u[1] = g0.y; a[s].u[2] = g0.z; a[s].u[3] = g0.w;
                a[s].u[4] = g1.x; a[s].u[5] = g1.y; a[s].u[6] = g1.z; a[s].u[7] = g1.w;
            }
        }

        // Keep all 16 loads issued before any WMMA: distinct destination
        // registers stay live, so the waits become staggered partial
        // s_wait_loadcnt values instead of draining to 0 before each WMMA.
        SCHED_BARRIER();

        // 4 independent WMMA accumulation chains (one per i-tile), then 4
        // independent dot-product partials (fills WMMA->VALU hazard slots).
        float ps[4];
        #pragma unroll
        for (int it = 0; it < 4; ++it) {
            v8f acc = {};
            acc = __builtin_amdgcn_wmma_f32_16x16x32_f16(
                    false, a[it * 2 + 0].v, false, bm[0].v, (short)0, acc, false, false);
            acc = __builtin_amdgcn_wmma_f32_16x16x32_f16(
                    false, a[it * 2 + 1].v, false, bm[1].v, (short)0, acc, false, false);
            float p = 0.0f;
            #pragma unroll
            for (int g = 0; g < 8; ++g) p += acc[g] * wyr[it * 8 + g];
            ps[it] = p;
        }
        float part = (ps[0] + ps[1]) + (ps[2] + ps[3]);

        float tot = part + __shfl_xor(part, 16, 32);   // add other i-half
        if (half == 0) out[(size_t)roi * CCH + c] = invr * tot;
    }
}

extern "C" void kernel_launch(void* const* d_in, const int* in_sizes, int n_in,
                              void* d_out, int out_size, void* d_ws, size_t ws_size,
                              hipStream_t stream) {
    const float* feature = (const float*)d_in[0];   // [8,64,64,64] f32
    const float* rois    = (const float*)d_in[1];   // [512,8,1,1,5] f32
    float*       out     = (float*)d_out;           // [4096,64,1,1] f32

    char* ws = (char*)d_ws;
    size_t off = 0;
    float* Wx = (float*)(ws + off);            off += (size_t)R_TOT * WW * 4;   // 1 MB
    float* Wy = (float*)(ws + off);            off += (size_t)R_TOT * HH * 4;   // 1 MB
    float* inv = (float*)(ws + off);           off += (size_t)R_TOT * 4;        // 16 KB
    int* counts = (int*)(ws + off);            off += 64;
    int* lists = (int*)(ws + off);             off += (size_t)NB * R_TOT * 4;   // 128 KB
    int* tileRoi = (int*)(ws + off);           off += (size_t)MAXT * 16 * 4;
    int* tileBatch = (int*)(ws + off);         off += (size_t)MAXT * 4;
    off = (off + 15) & ~(size_t)15;
    _Float16* feat16 = (_Float16*)(ws + off);  off += (size_t)NB * CCH * HH * WW * 2; // 4 MB

    const int nfeat = NB * CCH * HH * WW;      // 2,097,152
    const int n4 = nfeat / 4;

    hipLaunchKernelGGL(k_zero, dim3(1), dim3(32), 0, stream, counts);
    hipLaunchKernelGGL(k_cvt, dim3((n4 + 255) / 256), dim3(256), 0, stream,
                       feature, feat16, n4);
    hipLaunchKernelGGL(k_weights, dim3(R_TOT / 256), dim3(256), 0, stream,
                       rois, Wx, Wy, inv, counts, lists);
    hipLaunchKernelGGL(k_tiles, dim3(1), dim3(1), 0, stream,
                       counts, lists, tileRoi, tileBatch);
    hipLaunchKernelGGL(k_pool, dim3(MAXT), dim3(128), 0, stream,
                       feat16, Wx, Wy, inv, tileRoi, tileBatch, out);
}
